// GATNet_20804821581833
// MI455X (gfx1250) — matrix operand order repaired and checked by reading, not compiled, not run on hardware
//
#include <hip/hip_runtime.h>
#include <hip/hip_bf16.h>
#include <math.h>

// ---------------------------------------------------------------------------
// GAT (2-layer) for MI455X / gfx1250.
//  N=50000 nodes, F=128, L1: 8 heads x 64 ch, L2: 1 head x 64 ch, E=800000.
//  GEMMs: v_wmma_f32_16x16x32_f16 with the full weight panel staged in LDS
//  (pre-swizzled into B-fragment order -> ds_load_b128 per fragment), A
//  fragments reused across 4 N-tiles. Edge softmax/aggregate use L2 float
//  atomics (entire working set fits the 192 MB L2).
// ---------------------------------------------------------------------------

typedef __attribute__((ext_vector_type(16))) _Float16 v16h;
typedef __attribute__((ext_vector_type(8)))  float    v8f;

#define IN_F      128
#define L1_HEADS  8
#define L1_CH     64
#define L2_CH     64
#define NEG_SLOPE 0.2f
#define NT_GROUP  4      // N-tiles processed per A-fragment load

// Monotonic float <-> uint mapping so atomicMax(uint) == float max (handles negatives).
__device__ __forceinline__ unsigned fenc(float f) {
  unsigned u = __float_as_uint(f);
  return (u & 0x80000000u) ? ~u : (u | 0x80000000u);
}
__device__ __forceinline__ float fdec(unsigned e) {
  return (e & 0x80000000u) ? __uint_as_float(e & 0x7FFFFFFFu) : __uint_as_float(~e);
}

// ---------------------------------------------------------------------------
// WMMA GEMM: C[M,N] = A[M,K] @ B[K,N], f32 in/out, f16 multiply, f32 accum.
//  - whole B panel staged once per block into dynamic LDS, converted to f16
//    and laid out in WMMA B-fragment order: each lane's 16 halves contiguous
//    (32 B) -> 2x ds_load_b128 per fragment.
//  - one wave per 16-row M-tile; A fragment loaded once per (k-chunk, group)
//    and reused across NT_GROUP consecutive N-tile WMMAs.
//  - optional fused ELU on A (layer-2 input h = elu(emb)).
// Fragment layouts per CDNA5 ISA 7.12.2 (wave32).
// Requires: M%16==0, K%32==0, (N/16)%NT_GROUP==0.
// ---------------------------------------------------------------------------
template <bool ELU>
__global__ __launch_bounds__(256) void gemm_lds_wmma(
    const float* __restrict__ A, const float* __restrict__ B,
    float* __restrict__ C, int M, int N, int K)
{
  extern __shared__ _Float16 smemh[];   // K*N halves, fragment-ordered

  const int ntiles  = N >> 4;
  const int mtiles  = M >> 4;
  const int lane    = threadIdx.x & 31;
  const bool hi     = lane >= 16;
  const int  l15    = lane & 15;

  // ---- stage B panel into LDS in fragment order --------------------------
  // linear index u = (((kc*ntiles + nt)*32 + lane)*16 + i)
  // lane<16 : half i = B[kc*32 + i + (i>=8 ? 8 : 0)    ][nt*16 + lane   ]
  // lane>=16: half i = B[kc*32 + 8 + i + (i>=8 ? 8 : 0)][nt*16 + lane-16]
  const int total = K * N;
  for (int u = threadIdx.x; u < total; u += blockDim.x) {
    int i  = u & 15;
    int ln = (u >> 4) & 31;
    int ft = u >> 9;                 // kc*ntiles + nt
    int nt = ft % ntiles;
    int kc = ft / ntiles;
    int kk = (kc << 5) + ((ln >= 16) ? 8 : 0) + i + ((i >= 8) ? 8 : 0);
    int nn = (nt << 4) + (ln & 15);
    smemh[u] = (_Float16)B[(size_t)kk * N + nn];
  }
  __syncthreads();

  // ---- compute: one wave per M-tile --------------------------------------
  const int mt = (blockIdx.x * blockDim.x + threadIdx.x) >> 5;
  if (mt >= mtiles) return;          // after barrier; wave-uniform
  const int row = (mt << 4) + l15;   // A-matrix row for this lane
  const int kchunks = K >> 5;

  for (int nt0 = 0; nt0 < ntiles; nt0 += NT_GROUP) {
    v8f acc[NT_GROUP];
#pragma unroll
    for (int g = 0; g < NT_GROUP; ++g) acc[g] = (v8f){};

    for (int kci = 0; kci < kchunks; ++kci) {
      // A 16x32 f16 fragment: lane<16 holds K {kc..kc+7, kc+16..kc+23},
      //                       lane>=16 holds K {kc+8..kc+15, kc+24..kc+31}.
      const float* ap = A + (size_t)row * K + (kci << 5) + (hi ? 8 : 0);
      v16h a;
#pragma unroll
      for (int i = 0; i < 8; ++i) {
        float x0 = ap[i];
        float x1 = ap[i + 16];
        if (ELU) {
          x0 = x0 > 0.f ? x0 : (__expf(x0) - 1.f);
          x1 = x1 > 0.f ? x1 : (__expf(x1) - 1.f);
        }
        a[i]     = (_Float16)x0;
        a[i + 8] = (_Float16)x1;
      }
#pragma unroll
      for (int g = 0; g < NT_GROUP; ++g) {
        const v16h b = *(const v16h*)(smemh +
            (((size_t)(kci * ntiles + nt0 + g) * 32 + lane) << 4));
        acc[g] = __builtin_amdgcn_wmma_f32_16x16x32_f16(
            /*neg_a=*/false, a, /*neg_b=*/false, b,
            /*c_mod=*/(short)0, acc[g], /*reuse_a=*/false, /*reuse_b=*/false);
      }
    }

    // C/D layout: VGPR v -> row (hi ? v+8 : v), col = lane&15.
    const int rb = hi ? 8 : 0;
#pragma unroll
    for (int g = 0; g < NT_GROUP; ++g) {
      float* cp = C + (size_t)(mt << 4) * N + ((nt0 + g) << 4) + l15;
#pragma unroll
      for (int v = 0; v < 8; ++v) cp[(size_t)(rb + v) * N] = acc[g][v];
    }
  }
}

// ---------------------------------------------------------------------------
// Per-node attention halves: al_s[n,h] = <xw[n,h,:], a_src[h,:]>, same for dst.
// ---------------------------------------------------------------------------
__global__ void attn_logits(const float* __restrict__ xw,
                            const float* __restrict__ a_src,
                            const float* __restrict__ a_dst,
                            float* __restrict__ al_s, float* __restrict__ al_d,
                            int n, int heads, int ch)
{
  int t = blockIdx.x * blockDim.x + threadIdx.x;
  if (t >= n * heads) return;
  int node = t / heads, h = t - node * heads;
  const float* xr = xw + (size_t)node * heads * ch + (size_t)h * ch;
  const float* as = a_src + h * ch;
  const float* ad = a_dst + h * ch;
  float s = 0.f, d = 0.f;
#pragma unroll 8
  for (int c = 0; c < ch; ++c) {
    float v = xr[c];
    s += v * as[c];
    d += v * ad[c];
  }
  al_s[t] = s;
  al_d[t] = d;
}

// ---------------------------------------------------------------------------
// Pass 1: segment max of leaky_relu(al_s[src]+al_d[dst]) into m[dst] (uint enc).
// Edges e in [0,E) come from edge_index; e in [E, E+n) are self-loops.
// ---------------------------------------------------------------------------
__global__ void edge_max(const int* __restrict__ src, const int* __restrict__ dst,
                         const float* __restrict__ al_s, const float* __restrict__ al_d,
                         unsigned* __restrict__ m, int E, int n, int heads)
{
  int t = blockIdx.x * blockDim.x + threadIdx.x;
  int total = (E + n) * heads;
  if (t >= total) return;
  int e = t / heads, h = t - e * heads;
  int s_, d_;
  if (e < E) { s_ = src[e]; d_ = dst[e]; } else { s_ = d_ = e - E; }
  float a = al_s[(size_t)s_ * heads + h] + al_d[(size_t)d_ * heads + h];
  a = a >= 0.f ? a : NEG_SLOPE * a;
  atomicMax(&m[(size_t)d_ * heads + h], fenc(a));
}

// Pass 2: segment sum of exp(alpha - m[dst]) into s[dst].
__global__ void edge_sum(const int* __restrict__ src, const int* __restrict__ dst,
                         const float* __restrict__ al_s, const float* __restrict__ al_d,
                         const unsigned* __restrict__ m, float* __restrict__ sm,
                         int E, int n, int heads)
{
  int t = blockIdx.x * blockDim.x + threadIdx.x;
  int total = (E + n) * heads;
  if (t >= total) return;
  int e = t / heads, h = t - e * heads;
  int s_, d_;
  if (e < E) { s_ = src[e]; d_ = dst[e]; } else { s_ = d_ = e - E; }
  float a = al_s[(size_t)s_ * heads + h] + al_d[(size_t)d_ * heads + h];
  a = a >= 0.f ? a : NEG_SLOPE * a;
  float w = __expf(a - fdec(m[(size_t)d_ * heads + h]));
  atomicAdd(&sm[(size_t)d_ * heads + h], w);
}

// ---------------------------------------------------------------------------
// Pass 3: agg[dst,h,:] += coeff(e,h) * xw[src,h,:].  One wave32 per edge;
// coeff uniform per head, channels handled in coalesced 32-lane chunks.
// L2-resident scatter (agg + xw fit in the 192 MB L2).
// ---------------------------------------------------------------------------
__global__ void edge_aggregate(const int* __restrict__ src, const int* __restrict__ dst,
                               const float* __restrict__ al_s, const float* __restrict__ al_d,
                               const unsigned* __restrict__ m, const float* __restrict__ sm,
                               const float* __restrict__ xw, float* __restrict__ agg,
                               int E, int n, int heads, int ch)
{
  int wave = (blockIdx.x * blockDim.x + threadIdx.x) >> 5;
  int lane = threadIdx.x & 31;
  if (wave >= E + n) return;
  int s_, d_;
  if (wave < E) { s_ = src[wave]; d_ = dst[wave]; } else { s_ = d_ = wave - E; }
  for (int h = 0; h < heads; ++h) {
    float a = al_s[(size_t)s_ * heads + h] + al_d[(size_t)d_ * heads + h];
    a = a >= 0.f ? a : NEG_SLOPE * a;
    float coeff = __expf(a - fdec(m[(size_t)d_ * heads + h])) /
                  sm[(size_t)d_ * heads + h];
    const float* xr = xw  + (size_t)s_ * heads * ch + (size_t)h * ch;
    float*       ag = agg + (size_t)d_ * heads * ch + (size_t)h * ch;
    for (int c = lane; c < ch; c += 32)
      atomicAdd(&ag[c], coeff * xr[c]);
  }
}

// out[n, C] += b[C] (broadcast bias).
__global__ void add_bias(float* __restrict__ out, const float* __restrict__ b,
                         int n, int C)
{
  int t = blockIdx.x * blockDim.x + threadIdx.x;
  if (t >= n * C) return;
  out[t] += b[t % C];
}

// ---------------------------------------------------------------------------
extern "C" void kernel_launch(void* const* d_in, const int* in_sizes, int n_in,
                              void* d_out, int out_size, void* d_ws, size_t ws_size,
                              hipStream_t stream)
{
  const float* x   = (const float*)d_in[0];
  const int*   ei  = (const int*)d_in[1];
  const float* W1  = (const float*)d_in[2];
  const float* a1s = (const float*)d_in[3];
  const float* a1d = (const float*)d_in[4];
  const float* b1  = (const float*)d_in[5];
  const float* W2  = (const float*)d_in[6];
  const float* a2s = (const float*)d_in[7];
  const float* a2d = (const float*)d_in[8];
  const float* b2  = (const float*)d_in[9];

  const int n  = in_sizes[0] / IN_F;      // 50000
  const int E  = in_sizes[1] / 2;         // 800000
  const int C1 = L1_HEADS * L1_CH;        // 512
  const int C2 = L2_CH;                   // 64
  const int* src = ei;
  const int* dst = ei + E;

  // Output layout: logits [n*64] then emb [n*512] (reference tuple order).
  float* logits = (float*)d_out;
  float* emb    = logits + (size_t)n * C2;

  // Workspace carve-up (~122 MB of f32).
  float* p = (float*)d_ws;
  float*    xw1  = p;            p += (size_t)n * C1;
  float*    al1S = p;            p += (size_t)n * L1_HEADS;
  float*    al1D = p;            p += (size_t)n * L1_HEADS;
  unsigned* m1   = (unsigned*)p; p += (size_t)n * L1_HEADS;
  float*    s1   = p;            p += (size_t)n * L1_HEADS;   // contiguous after m1
  float*    xw2  = p;            p += (size_t)n * C2;
  float*    al2S = p;            p += n;
  float*    al2D = p;            p += n;
  unsigned* m2   = (unsigned*)p; p += n;
  float*    s2   = p;            p += n;                      // contiguous after m2
  (void)ws_size; (void)n_in;

  // Zero accumulators (graph-capture-legal). m=0u decodes correctly since
  // every node has a self-loop; s=0 is the sum identity.
  hipMemsetAsync(d_out, 0, (size_t)out_size * sizeof(float), stream);
  hipMemsetAsync(m1, 0, (size_t)n * L1_HEADS * 2 * sizeof(float), stream); // m1+s1
  hipMemsetAsync(m2, 0, (size_t)n * 2 * sizeof(float), stream);            // m2+s2

  const int TPB = 256;                     // 8 waves per block
  const int mtiles = n >> 4;               // 3125
  const int gemm_blocks = (mtiles + 7) / 8;

  // ---------------- Layer 1 (8 heads x 64 ch, concat) ----------------
  {
    size_t lds = (size_t)IN_F * C1 * sizeof(_Float16);   // 128 KB weight panel
    gemm_lds_wmma<false><<<gemm_blocks, TPB, lds, stream>>>(x, W1, xw1, n, C1, IN_F);
  }
  attn_logits<<<(n * L1_HEADS + TPB - 1) / TPB, TPB, 0, stream>>>(
      xw1, a1s, a1d, al1S, al1D, n, L1_HEADS, L1_CH);
  {
    int tot = (E + n) * L1_HEADS;
    edge_max<<<(tot + TPB - 1) / TPB, TPB, 0, stream>>>(src, dst, al1S, al1D, m1, E, n, L1_HEADS);
    edge_sum<<<(tot + TPB - 1) / TPB, TPB, 0, stream>>>(src, dst, al1S, al1D, m1, s1, E, n, L1_HEADS);
  }
  edge_aggregate<<<((unsigned)(E + n) * 32u + TPB - 1) / TPB, TPB, 0, stream>>>(
      src, dst, al1S, al1D, m1, s1, xw1, emb, E, n, L1_HEADS, L1_CH);
  add_bias<<<((unsigned)n * C1 + TPB - 1) / TPB, TPB, 0, stream>>>(emb, b1, n, C1);

  // ---------------- Layer 2 (1 head x 64 ch, mean == identity) ----------------
  {
    size_t lds = (size_t)C1 * C2 * sizeof(_Float16);     // 64 KB weight panel
    gemm_lds_wmma<true><<<gemm_blocks, TPB, lds, stream>>>(emb, W2, xw2, n, C2, C1);
  }
  attn_logits<<<(n + TPB - 1) / TPB, TPB, 0, stream>>>(
      xw2, a2s, a2d, al2S, al2D, n, 1, L2_CH);
  {
    int tot = E + n;
    edge_max<<<(tot + TPB - 1) / TPB, TPB, 0, stream>>>(src, dst, al2S, al2D, m2, E, n, 1);
    edge_sum<<<(tot + TPB - 1) / TPB, TPB, 0, stream>>>(src, dst, al2S, al2D, m2, s2, E, n, 1);
  }
  edge_aggregate<<<((unsigned)(E + n) * 32u + TPB - 1) / TPB, TPB, 0, stream>>>(
      src, dst, al2S, al2D, m2, s2, xw2, logits, E, n, 1, L2_CH);
  add_bias<<<((unsigned)n * C2 + TPB - 1) / TPB, TPB, 0, stream>>>(logits, b2, n, C2);
}
